// MemoryAccess_36558761624036
// MI455X (gfx1250) — matrix operand order repaired and checked by reading, not compiled, not run on hardware
//
#include <hip/hip_runtime.h>
#include <math.h>

// ---------------------------------------------------------------------------
// DNC MemoryAccess forward for MI455X (gfx1250, wave32, WMMA).
//
// Shapes: B=16 T=32 D=512 N=1024 W=64 R=4 NW=2.
// Memory-bound: link matrix (B,NW,N,N) dominates (~400MB total traffic
// ~17us @ 23.3TB/s). GEMM-shaped pieces (dense proj, cosine sims, fw/bw
// products against new_link) use v_wmma_f32_16x16x32_f16 with f32 accum.
//
// Input order (setup_inputs insertion order, params dict flattened in
// insertion order, 'w' then 'b' per dense):
//  0 inputs(B,T,D) 1 memory(B,N,W) 2 read_weights(B,1,R,N)
//  3 write_weights(B,1,NW,N) 4 link(B,NW,N,N) 5 precedence(B,NW,N)
//  6 usage(B,N) 7.. params: wv.w/b ev.w/b fg.w/b ag.w/b wg.w/b rm.w/b
//                            wk.w/b ws.w/b rk.w/b rs.w/b
// Output (concat): read_words(B,R,W), new_memory(B,N,W), new_rw(B,1,R,N),
//  fww(B,1,NW,N), new_link(B,NW,N,N), new_prec(B,NW,N), new_usage(B,N)
// ---------------------------------------------------------------------------

typedef __attribute__((ext_vector_type(16))) _Float16 v16h;
typedef __attribute__((ext_vector_type(8)))  float    v8f;

constexpr int cB = 16, cT = 32, cD = 512, cN = 1024, cW = 64, cR = 4, cNW = 2;
constexpr int TOK  = cB * cT;   // 512
constexpr int NOUT = 674;       // total dense output units
constexpr int NPAD = 688;       // 43*16
constexpr float kEPS = 1e-6f;

// ---------------- WMMA helpers (contiguous k = 16h + s layouts) -----------------------

__device__ inline v8f vz8() { v8f z = {0.f,0.f,0.f,0.f,0.f,0.f,0.f,0.f}; return z; }

__device__ inline v8f wmma32(v16h a, v16h b, v8f c) {
  // v_wmma_f32_16x16x32_f16 : D = A(16x32 f16) x B(32x16 f16) + C(16x16 f32)
  return __builtin_amdgcn_wmma_f32_16x16x32_f16(false, a, false, b,
                                                (short)0, c, false, false);
}

// A-matrix 16x32 f16 layout: lane m=lane%16.
// lane<16 : elems 0..15 <- K k0..k0+15
// lane>=16: elems 0..15 <- K k0+16..k0+31
__device__ inline v16h load_a_f32(const float* __restrict__ A, int ld, int m,
                                  int k0, int laneHi) {
  v16h r;
  const float* p = A + (size_t)m * ld + k0 + 16 * laneHi;
#pragma unroll
  for (int e = 0; e < 8; ++e) r[e] = (_Float16)p[e];
#pragma unroll
  for (int e = 0; e < 8; ++e) r[8 + e] = (_Float16)p[8 + e];
  return r;
}
__device__ inline v16h load_a_f16(const _Float16* __restrict__ A, int ld, int m,
                                  int k0, int laneHi) {
  v16h r;
  const _Float16* p = A + (size_t)m * ld + k0 + 16 * laneHi;
#pragma unroll
  for (int e = 0; e < 8; ++e) r[e] = p[e];
#pragma unroll
  for (int e = 0; e < 8; ++e) r[8 + e] = p[8 + e];
  return r;
}
// B-matrix 32x16 f16 layout: lane n=lane%16, K-base = k0 + 16*(lane>=16),
// elems e <- B[kbase+e][n]. Source here is n-major ("B transposed") f16.
__device__ inline v16h load_bT_f16(const _Float16* __restrict__ Bt, int ld,
                                   int n, int k0, int laneHi) {
  v16h r;
  const _Float16* p = Bt + (size_t)n * ld + k0 + 16 * laneHi;
#pragma unroll
  for (int e = 0; e < 16; ++e) r[e] = p[e];
  return r;
}
// B from a k-major f32 matrix (column gather; 16 lanes cover 64B per row).
__device__ inline v16h load_b_colf32(const float* __restrict__ Bm, int ld,
                                     int n, int k0, int laneHi) {
  v16h r;
  const float* p = Bm + (size_t)(k0 + 16 * laneHi) * ld + n;
#pragma unroll
  for (int e = 0; e < 16; ++e) r[e] = (_Float16)p[(size_t)e * ld];
  return r;
}

// ---------------- Kernels ---------------------------------------------------

struct WPtrs { const float* w[10]; const float* b[10]; };

// Concatenate the 10 dense weight matrices transposed into whT[NPAD][512] f16
// (row = output unit, col = k) + concatenated bias. Pad rows zeroed.
__global__ void k_stage_weights(WPtrs wp, _Float16* __restrict__ whT,
                                float* __restrict__ bcat) {
  const int starts[11] = {0,128,256,260,262,264,284,412,414,670,674};
  int j = blockIdx.x;
  int src = -1, lj = 0, dout = 0;
#pragma unroll
  for (int s = 0; s < 10; ++s)
    if (j >= starts[s] && j < starts[s + 1]) { src = s; lj = j - starts[s]; dout = starts[s+1]-starts[s]; }
  for (int k = threadIdx.x; k < cD; k += blockDim.x)
    whT[(size_t)j * cD + k] = (_Float16)(src >= 0 ? wp.w[src][(size_t)k * dout + lj] : 0.f);
  if (threadIdx.x == 0) bcat[j] = (src >= 0 ? wp.b[src][lj] : 0.f);
}

// y[512][688] = x(512x512) @ Wcat + b, via WMMA. One wave per 16x16 tile.
__global__ void k_gemm(const float* __restrict__ x, const _Float16* __restrict__ whT,
                       const float* __restrict__ bcat, float* __restrict__ y) {
  int lane = threadIdx.x & 31, wave = threadIdx.x >> 5;
  int laneHi = lane >> 4, lane16 = lane & 15;
  int tile = blockIdx.x * 8 + wave;        // 32*43 = 1376 tiles, 172 blocks
  int mt = tile / 43, nt = tile % 43;
  int m = mt * 16 + lane16, n = nt * 16 + lane16;
  v8f acc = vz8();
  for (int k0 = 0; k0 < cD; k0 += 32) {
    v16h a = load_a_f32(x, cD, m, k0, laneHi);
    v16h b = load_bT_f16(whT, cD, n, k0, laneHi);
    acc = wmma32(a, b, acc);
  }
  float bb = bcat[n];
#pragma unroll
  for (int v = 0; v < 8; ++v) {
    int row = mt * 16 + v + 8 * laneHi;    // D: row = v + 8*(lane>=16)
    y[(size_t)row * NPAD + n] = acc[v] + bb;
  }
}

// In-place activations: sigmoid [128,264); softplus {412,413,670..673};
// softmax per read_mode group of 5 (cols 264..283).
__global__ void k_act(float* __restrict__ y) {
  int tok = blockIdx.x;
  float* row = y + (size_t)tok * NPAD;
  for (int j = threadIdx.x; j < NOUT; j += blockDim.x) {
    float v = row[j];
    if (j >= 128 && j < 264)
      row[j] = 1.f / (1.f + __expf(-v));
    else if (j == 412 || j == 413 || (j >= 670 && j < 674))
      row[j] = (v > 20.f) ? v : log1pf(__expf(v));
  }
  __syncthreads();
  if (threadIdx.x < 4) {
    int base = 264 + threadIdx.x * 5;
    float m = row[base];
#pragma unroll
    for (int q = 1; q < 5; ++q) m = fmaxf(m, row[base + q]);
    float e[5], s = 0.f;
#pragma unroll
    for (int q = 0; q < 5; ++q) { e[q] = __expf(row[base + q] - m); s += e[q]; }
#pragma unroll
    for (int q = 0; q < 5; ++q) row[base + q] = e[q] / s;
  }
}

// Per-batch sums over T for the 284 "small" dense columns.
__global__ void k_smallsum(const float* __restrict__ y, float* __restrict__ smalls) {
  int b = blockIdx.x, c = threadIdx.x;
  if (c < 284) {
    float s = 0.f;
    for (int tt = 0; tt < cT; ++tt) s += y[(size_t)(b * cT + tt) * NPAD + c];
    smalls[b * 288 + c] = s;
  }
}

// Stage memory rows as f16 + row norms. One 64-thread block per (b,n) row.
__global__ void k_memstage(const float* __restrict__ mem, _Float16* __restrict__ memh,
                           float* __restrict__ mn) {
  int rowi = blockIdx.x, k = threadIdx.x;
  float v = mem[(size_t)rowi * cW + k];
  memh[(size_t)rowi * cW + k] = (_Float16)v;
  __shared__ float rb[64];
  rb[k] = v * v; __syncthreads();
  for (int s = 32; s > 0; s >>= 1) { if (k < s) rb[k] += rb[k + s]; __syncthreads(); }
  if (k == 0) mn[rowi] = sqrtf(rb[0]);
}

// Stage per-token key matrix (16 rows padded: 0-1 write keys, 2-5 read keys)
// as f16 + key norms. grid (token, row), 64 threads.
__global__ void k_keystage(const float* __restrict__ y, _Float16* __restrict__ kh,
                           float* __restrict__ kn) {
  int tok = blockIdx.x, r = blockIdx.y, k = threadIdx.x;
  float v = 0.f;
  if (r < 2)       v = y[(size_t)tok * NPAD + 284 + r * cW + k];
  else if (r < 6)  v = y[(size_t)tok * NPAD + 414 + (r - 2) * cW + k];
  kh[((size_t)tok * 16 + r) * cW + k] = (_Float16)v;
  __shared__ float rb[64];
  rb[k] = v * v; __syncthreads();
  for (int s = 32; s > 0; s >>= 1) { if (k < s) rb[k] += rb[k + s]; __syncthreads(); }
  if (k == 0 && r < 6) kn[tok * 6 + r] = sqrtf(rb[0]);
}

// Cosine similarity via WMMA + per-row softmax. One block (4 waves) per token.
__global__ void k_cos(const _Float16* __restrict__ kh, const _Float16* __restrict__ memh,
                      const float* __restrict__ mn, const float* __restrict__ kn,
                      const float* __restrict__ y,
                      float* __restrict__ wcw_tok, float* __restrict__ rcw_tok) {
  __shared__ float sim[6 * cN];
  __shared__ float red[128];
  int tok = blockIdx.x, bm = tok % cB;     // mem_tok = memory[tok % B]
  int lane = threadIdx.x & 31, wave = threadIdx.x >> 5;
  int laneHi = lane >> 4, lane16 = lane & 15;
  const _Float16* Ak = kh + (size_t)tok * 16 * cW;
  v16h a0 = load_a_f16(Ak, cW, lane16, 0, laneHi);
  v16h a1 = load_a_f16(Ak, cW, lane16, 32, laneHi);
  const _Float16* Bm = memh + (size_t)bm * cN * cW;
  for (int t16 = 0; t16 < 16; ++t16) {
    int n = (wave * 16 + t16) * 16 + lane16;
    v16h b0 = load_bT_f16(Bm, cW, n, 0, laneHi);
    v16h b1 = load_bT_f16(Bm, cW, n, 32, laneHi);
    v8f acc = vz8();
    acc = wmma32(a0, b0, acc);
    acc = wmma32(a1, b1, acc);
    if (laneHi == 0) {
#pragma unroll
      for (int v = 0; v < 6; ++v) sim[v * cN + n] = acc[v];  // rows m=v (<6)
    }
  }
  __syncthreads();
  for (int r = 0; r < 6; ++r) {
    float strength = (r < 2) ? y[(size_t)tok * NPAD + 412 + r]
                             : y[(size_t)tok * NPAD + 670 + (r - 2)];
    float knv = kn[tok * 6 + r];
    float lv[8], lmax = -1e30f;
#pragma unroll
    for (int q = 0; q < 8; ++q) {
      int n = threadIdx.x + q * 128;
      float l = sim[r * cN + n] * strength / (knv * mn[(size_t)bm * cN + n] + kEPS);
      lv[q] = l; lmax = fmaxf(lmax, l);
    }
    red[threadIdx.x] = lmax; __syncthreads();
    for (int s = 64; s > 0; s >>= 1) { if (threadIdx.x < s) red[threadIdx.x] = fmaxf(red[threadIdx.x], red[threadIdx.x + s]); __syncthreads(); }
    lmax = red[0]; __syncthreads();
    float lsum = 0.f;
#pragma unroll
    for (int q = 0; q < 8; ++q) { lv[q] = __expf(lv[q] - lmax); lsum += lv[q]; }
    red[threadIdx.x] = lsum; __syncthreads();
    for (int s = 64; s > 0; s >>= 1) { if (threadIdx.x < s) red[threadIdx.x] += red[threadIdx.x + s]; __syncthreads(); }
    float inv = 1.f / red[0]; __syncthreads();
#pragma unroll
    for (int q = 0; q < 8; ++q) {
      int n = threadIdx.x + q * 128;
      float o = lv[q] * inv;
      if (r < 2) wcw_tok[((size_t)tok * cNW + r) * cN + n] = o;
      else       rcw_tok[((size_t)tok * cR + (r - 2)) * cN + n] = o;
    }
    __syncthreads();
  }
}

// tsum over T of the token-wise content weights.
__global__ void k_cwsum(const float* __restrict__ wcw_tok, const float* __restrict__ rcw_tok,
                        float* __restrict__ wcw_s, float* __restrict__ rcw_s) {
  int idx = blockIdx.x * 256 + threadIdx.x;       // 16*6*1024 = 98304
  int n = idx & 1023, rr = (idx >> 10) % 6, b = idx / 6144;
  float s = 0.f;
  if (rr < 4) {
    for (int tt = 0; tt < cT; ++tt) s += rcw_tok[((size_t)(b * cT + tt) * cR + rr) * cN + n];
    rcw_s[((size_t)b * cR + rr) * cN + n] = s;
  } else {
    int w = rr - 4;
    for (int tt = 0; tt < cT; ++tt) s += wcw_tok[((size_t)(b * cT + tt) * cNW + w) * cN + n];
    wcw_s[((size_t)b * cNW + w) * cN + n] = s;
  }
}

// Allocation weighting (bitonic argsort + product scan), fww, usage, prec.
// One 1024-thread block per batch.
__global__ void __launch_bounds__(1024)
k_alloc(const float* __restrict__ usage_in, const float* __restrict__ prec_in,
        const float* __restrict__ smalls, const float* __restrict__ wcw_s,
        const float* __restrict__ rcw_s, float* __restrict__ o_fww,
        float* __restrict__ o_prec, float* __restrict__ o_usage) {
  __shared__ float uArr[1024], key[1024], scan[1024], alA[1024], alB[1024], red[1024];
  __shared__ int sidx[1024];
  int b = blockIdx.x, t = threadIdx.x;
  const float* sm = smalls + b * 288;
  float usage0 = usage_in[(size_t)b * cN + t];
  uArr[t] = usage0;
  float gate[2] = { sm[260] * sm[262], sm[261] * sm[263] };   // ag_s * wg_s
  for (int it = 0; it < cNW; ++it) {
    __syncthreads();
    float uu = kEPS + (1.f - kEPS) * uArr[t];
    key[t] = 1.f - uu; sidx[t] = t;
    __syncthreads();
    for (int size = 2; size <= 1024; size <<= 1) {            // descending sort
      for (int stride = size >> 1; stride > 0; stride >>= 1) {
        int partner = t ^ stride;
        if (partner > t) {
          bool desc = ((t & size) == 0);
          float k1 = key[t], k2 = key[partner];
          if (desc ? (k1 < k2) : (k1 > k2)) {
            key[t] = k2; key[partner] = k1;
            int tmp = sidx[t]; sidx[t] = sidx[partner]; sidx[partner] = tmp;
          }
        }
        __syncthreads();
      }
    }
    scan[t] = (t == 0) ? 1.f : (1.f - key[t - 1]);            // excl cumprod src
    __syncthreads();
    for (int off = 1; off < 1024; off <<= 1) {
      float v = scan[t]; if (t >= off) v *= scan[t - off];
      __syncthreads(); scan[t] = v; __syncthreads();
    }
    float s_alloc = key[t] * scan[t];
    float* dst = (it == 0) ? alA : alB;
    dst[sidx[t]] = s_alloc;                                   // inverse scatter
    __syncthreads();
    uArr[t] = uArr[t] + (1.f - uArr[t]) * gate[it] * dst[t];
  }
  __syncthreads();
  float ag0 = sm[260], ag1 = sm[261], wg0 = sm[262], wg1 = sm[263];
  float f0 = wg0 * (ag0 * alA[t] + (1.f - ag0) * wcw_s[((size_t)b * cNW + 0) * cN + t]);
  float f1 = wg1 * (ag1 * alB[t] + (1.f - ag1) * wcw_s[((size_t)b * cNW + 1) * cN + t]);
  o_fww[((size_t)b * cNW + 0) * cN + t] = f0;
  o_fww[((size_t)b * cNW + 1) * cN + t] = f1;
  float agg = 1.f - (1.f - f0) * (1.f - f1);
  float u_after = usage0 + (1.f - usage0) * agg;
  float psi = 1.f;
#pragma unroll
  for (int r = 0; r < cR; ++r)
    psi *= (1.f - sm[256 + r] * rcw_s[((size_t)b * cR + r) * cN + t]);
  o_usage[(size_t)b * cN + t] = u_after * psi;
  red[t] = f0; __syncthreads();
  for (int s = 512; s > 0; s >>= 1) { if (t < s) red[t] += red[t + s]; __syncthreads(); }
  float S0 = red[0]; __syncthreads();
  red[t] = f1; __syncthreads();
  for (int s = 512; s > 0; s >>= 1) { if (t < s) red[t] += red[t + s]; __syncthreads(); }
  float S1 = red[0]; __syncthreads();
  o_prec[((size_t)b * cNW + 0) * cN + t] = (1.f - S0) * prec_in[((size_t)b * cNW + 0) * cN + t] + f0;
  o_prec[((size_t)b * cNW + 1) * cN + t] = (1.f - S1) * prec_in[((size_t)b * cNW + 1) * cN + t] + f1;
}

// new_memory = memory * prod_w(1 - fww_w*ev_w) + sum_w fww_w*wv_w
__global__ void k_newmem(const float* __restrict__ mem_in, const float* __restrict__ o_fww,
                         const float* __restrict__ smalls, float* __restrict__ o_new_mem) {
  size_t idx = (size_t)blockIdx.x * 256 + threadIdx.x;     // 16*1024*64
  int v = idx & 63, n = (idx >> 6) & 1023, b = idx >> 16;
  const float* sm = smalls + b * 288;
  float f0 = o_fww[((size_t)b * cNW + 0) * cN + n];
  float f1 = o_fww[((size_t)b * cNW + 1) * cN + n];
  float m = mem_in[idx];
  o_new_mem[idx] = m * (1.f - f0 * sm[128 + v]) * (1.f - f1 * sm[192 + v])
                 + f0 * sm[v] + f1 * sm[64 + v];
}

// new_link[i][j] = (1 - f_i - f_j)*L + f_i*p_j, zero diagonal.  Streamed.
__global__ void k_link(const float* __restrict__ link_in, const float* __restrict__ prec_in,
                       const float* __restrict__ o_fww, float* __restrict__ o_link) {
  int bz = blockIdx.x >> 4;          // b*NW + w  (0..31)
  int rt = blockIdx.x & 15;          // 64-row tile
  int j0 = threadIdx.x * 4;
  const float* fv = o_fww + (size_t)bz * cN;
  const float* pv = prec_in + (size_t)bz * cN;
  float4 fj = *(const float4*)(fv + j0);
  float4 pj = *(const float4*)(pv + j0);
  const float* Ls = link_in + (size_t)bz * cN * cN;
  float* Ld = o_link + (size_t)bz * cN * cN;
  for (int i = rt * 64; i < rt * 64 + 64; ++i) {
    float fi = fv[i];                                   // uniform -> scalar
    float om = 1.f - fi;
    const float4 L = *(const float4*)(Ls + (size_t)i * cN + j0);
    float4 nl;
    nl.x = (om - fj.x) * L.x + fi * pj.x;
    nl.y = (om - fj.y) * L.y + fi * pj.y;
    nl.z = (om - fj.z) * L.z + fi * pj.z;
    nl.w = (om - fj.w) * L.w + fi * pj.w;
    if (i == j0)     nl.x = 0.f;
    if (i == j0 + 1) nl.y = 0.f;
    if (i == j0 + 2) nl.z = 0.f;
    if (i == j0 + 3) nl.w = 0.f;
    *(float4*)(Ld + (size_t)i * cN + j0) = nl;
    __builtin_prefetch(Ls + (size_t)(i + 1) * cN + j0, 0, 0);  // global_prefetch
  }
}

// Stage prev_r padded to 16 rows, f16: ph[b][16][N].
__global__ void k_ph(const float* __restrict__ rw_in, _Float16* __restrict__ ph) {
  int idx = blockIdx.x * 256 + threadIdx.x;   // 16*16*1024
  int n = idx & 1023, r = (idx >> 10) & 15, b = idx >> 14;
  float v = (r < cR) ? rw_in[((size_t)b * cR + r) * cN + n] : 0.f;
  ph[idx] = (_Float16)v;
}

// fw[b,r,w,m] = sum_n new_link[b,w,m,n] * prev_r[b,r,n]   (WMMA, A=link rows)
__global__ void k_fw(const float* __restrict__ o_link, const _Float16* __restrict__ ph,
                     float* __restrict__ fwb) {
  int lane = threadIdx.x & 31, wave = threadIdx.x >> 5;
  int laneHi = lane >> 4, lane16 = lane & 15;
  int tile = blockIdx.x * 8 + wave;           // 32 mats * 64 m-tiles = 2048
  int b = tile >> 7, w = (tile >> 6) & 1, mt = tile & 63;
  const float* A = o_link + (size_t)(b * cNW + w) * cN * cN;
  const _Float16* Bt = ph + (size_t)b * 16 * cN;   // B[k][r] = ph[b][r][k]
  int m = mt * 16 + lane16;
  v8f acc = vz8();
  for (int k0 = 0; k0 < cN; k0 += 32) {
    v16h a  = load_a_f32(A, cN, m, k0, laneHi);
    v16h bb = load_bT_f16(Bt, cN, lane16, k0, laneHi);
    acc = wmma32(a, bb, acc);
  }
  int r = lane16;                              // D column = r (only 0..3 valid)
  if (r < cR) {
#pragma unroll
    for (int v = 0; v < 8; ++v) {
      int mrow = mt * 16 + v + 8 * laneHi;
      fwb[(((size_t)b * cR + r) * cNW + w) * cN + mrow] = acc[v];
    }
  }
}

// bw[b,r,w,m] = sum_n prev_r[b,r,n] * new_link[b,w,n,m]   (WMMA, A=prev_r)
__global__ void k_bw(const float* __restrict__ o_link, const _Float16* __restrict__ ph,
                     float* __restrict__ bwb) {
  int lane = threadIdx.x & 31, wave = threadIdx.x >> 5;
  int laneHi = lane >> 4, lane16 = lane & 15;
  int tile = blockIdx.x * 8 + wave;
  int b = tile >> 7, w = (tile >> 6) & 1, mt = tile & 63;
  const _Float16* Ap = ph + (size_t)b * 16 * cN;
  const float* Bm = o_link + (size_t)(b * cNW + w) * cN * cN;
  int mcol = mt * 16 + lane16;
  v8f acc = vz8();
  for (int k0 = 0; k0 < cN; k0 += 32) {
    v16h a  = load_a_f16(Ap, cN, lane16, k0, laneHi);
    v16h bb = load_b_colf32(Bm, cN, mcol, k0, laneHi);
    acc = wmma32(a, bb, acc);
  }
  if (laneHi == 0) {                           // D row r = v (+8*laneHi); need r<4
#pragma unroll
    for (int v = 0; v < cR; ++v)
      bwb[(((size_t)b * cR + v) * cNW + w) * cN + mcol] = acc[v];
  }
}

// new_rw and read_words. One block per (b,r).
__global__ void k_rw(const float* __restrict__ smalls, const float* __restrict__ rcw_s,
                     const float* __restrict__ fwb, const float* __restrict__ bwb,
                     const float* __restrict__ o_new_mem,
                     float* __restrict__ o_new_rw, float* __restrict__ o_read_words) {
  __shared__ float rwrow[cN];
  __shared__ float pbuf[4][cW];
  int b = blockIdx.x >> 2, r = blockIdx.x & 3;
  const float* sm = smalls + b * 288 + 264 + r * 5;   // rm_s group for r
  float cm = sm[0], fm0 = sm[1], fm1 = sm[2], bm0 = sm[3], bm1 = sm[4];
#pragma unroll
  for (int q = 0; q < 4; ++q) {
    int n = threadIdx.x + q * 256;
    size_t rb = ((size_t)b * cR + r) * cNW;
    float val = cm * rcw_s[((size_t)b * cR + r) * cN + n]
              + fm0 * fwb[(rb + 0) * cN + n] + fm1 * fwb[(rb + 1) * cN + n]
              + bm0 * bwb[(rb + 0) * cN + n] + bm1 * bwb[(rb + 1) * cN + n];
    rwrow[n] = val;
    o_new_rw[((size_t)b * cR + r) * cN + n] = val;
  }
  __syncthreads();
  int v = threadIdx.x & 63, part = threadIdx.x >> 6;
  float acc = 0.f;
  for (int n = part * 256; n < part * 256 + 256; ++n)
    acc += rwrow[n] * o_new_mem[((size_t)b * cN + n) * cW + v];
  pbuf[part][v] = acc; __syncthreads();
  if (part == 0)
    o_read_words[((size_t)b * cR + r) * cW + v] =
        pbuf[0][v] + pbuf[1][v] + pbuf[2][v] + pbuf[3][v];
}

// ---------------- Host side -------------------------------------------------

static inline size_t align256(size_t x) { return (x + 255) & ~(size_t)255; }

extern "C" void kernel_launch(void* const* d_in, const int* in_sizes, int n_in,
                              void* d_out, int out_size, void* d_ws, size_t ws_size,
                              hipStream_t stream) {
  const float* in_x     = (const float*)d_in[0];
  const float* in_mem   = (const float*)d_in[1];
  const float* in_rw    = (const float*)d_in[2];   // (B,1,R,N); prev_r = [:, -1]
  const float* in_prec  = (const float*)d_in[5];
  const float* in_usage = (const float*)d_in[6];
  WPtrs wp;
  for (int i = 0; i < 10; ++i) { wp.w[i] = (const float*)d_in[7 + 2 * i];
                                 wp.b[i] = (const float*)d_in[8 + 2 * i]; }

  // workspace layout
  char* wsb = (char*)d_ws; size_t off = 0;
  auto wsalloc = [&](size_t bytes) -> void* { void* p = wsb + off; off = align256(off + bytes); return p; };
  _Float16* whT    = (_Float16*)wsalloc((size_t)NPAD * cD * 2);
  float*    bcat   = (float*)wsalloc((size_t)NPAD * 4);
  float*    y      = (float*)wsalloc((size_t)TOK * NPAD * 4);
  float*    smalls = (float*)wsalloc((size_t)cB * 288 * 4);
  _Float16* memh   = (_Float16*)wsalloc((size_t)cB * cN * cW * 2);
  float*    mn     = (float*)wsalloc((size_t)cB * cN * 4);
  _Float16* kh     = (_Float16*)wsalloc((size_t)TOK * 16 * cW * 2);
  float*    kn     = (float*)wsalloc((size_t)TOK * 6 * 4);
  float*    wcw_tok= (float*)wsalloc((size_t)TOK * cNW * cN * 4);
  float*    rcw_tok= (float*)wsalloc((size_t)TOK * cR * cN * 4);
  float*    wcw_s  = (float*)wsalloc((size_t)cB * cNW * cN * 4);
  float*    rcw_s  = (float*)wsalloc((size_t)cB * cR * cN * 4);
  _Float16* ph     = (_Float16*)wsalloc((size_t)cB * 16 * cN * 2);
  float*    fwb    = (float*)wsalloc((size_t)cB * cR * cNW * cN * 4);
  float*    bwb    = (float*)wsalloc((size_t)cB * cR * cNW * cN * 4);
  (void)ws_size; (void)in_sizes; (void)n_in; (void)out_size;

  // output layout (reference return-tuple order, flat concat)
  float* out = (float*)d_out;
  float* o_read_words = out;                                   // B*R*W
  float* o_new_mem    = o_read_words + cB * cR * cW;           // B*N*W
  float* o_new_rw     = o_new_mem + (size_t)cB * cN * cW;      // B*1*R*N
  float* o_fww        = o_new_rw + (size_t)cB * cR * cN;       // B*1*NW*N
  float* o_link       = o_fww + (size_t)cB * cNW * cN;         // B*NW*N*N
  float* o_prec       = o_link + (size_t)cB * cNW * cN * cN;   // B*NW*N
  float* o_usage      = o_prec + (size_t)cB * cNW * cN;        // B*N

  k_stage_weights<<<NPAD, 256, 0, stream>>>(wp, whT, bcat);
  k_gemm<<<(32 * 43) / 8, 256, 0, stream>>>(in_x, whT, bcat, y);
  k_act<<<TOK, 256, 0, stream>>>(y);
  k_smallsum<<<cB, 288, 0, stream>>>(y, smalls);
  k_memstage<<<cB * cN, 64, 0, stream>>>(in_mem, memh, mn);
  k_keystage<<<dim3(TOK, 16), 64, 0, stream>>>(y, kh, kn);
  k_cos<<<TOK, 128, 0, stream>>>(kh, memh, mn, kn, y, wcw_tok, rcw_tok);
  k_cwsum<<<(cB * 6 * cN) / 256, 256, 0, stream>>>(wcw_tok, rcw_tok, wcw_s, rcw_s);
  k_alloc<<<cB, 1024, 0, stream>>>(in_usage, in_prec, smalls, wcw_s, rcw_s,
                                   o_fww, o_prec, o_usage);
  k_newmem<<<(cB * cN * cW) / 256, 256, 0, stream>>>(in_mem, o_fww, smalls, o_new_mem);
  k_link<<<cB * cNW * 16, 256, 0, stream>>>((const float*)d_in[4], in_prec, o_fww, o_link);
  k_ph<<<(cB * 16 * cN) / 256, 256, 0, stream>>>(in_rw, ph);
  k_fw<<<(cB * cNW * 64) / 8, 256, 0, stream>>>(o_link, ph, fwb);
  k_bw<<<(cB * cNW * 64) / 8, 256, 0, stream>>>(o_link, ph, bwb);
  k_rw<<<cB * cR, 256, 0, stream>>>(smalls, rcw_s, fwb, bwb, o_new_mem,
                                    o_new_rw, o_read_words);
}
